// BackgroundForegroundNeRF_45835890983143
// MI455X (gfx1250) — compile-verified, matrix operand order checked by reading, not run on hardware
//
#include <hip/hip_runtime.h>
#include <stdint.h>

// ---------------------------------------------------------------------------
// BackgroundForegroundNeRF fused inference for MI455X (gfx1250).
// Feature-major formulation: every layer computes h^T = W x act^T with
//   A operand = weights (pre-packed A-fragments in d_ws),
//   B operand = activations.
// Key property: the WMMA D (C-layout) of one layer converts to the next
// layer's B fragment with lane-LOCAL packed f16 converts only -- no LDS, no
// cross-lane shuffles between hidden layers. ReLU is applied on packed f16
// pairs with a single v_pk_max_num_f16 per dword. LDS is used only at the
// sigma-net -> color-net junction and for the scalar epilogue.
// Each wave owns 32 points (two 16-point B-tiles sharing every weight frag).
// ---------------------------------------------------------------------------

typedef __attribute__((ext_vector_type(16))) _Float16 v16h;
typedef __attribute__((ext_vector_type(8)))  float    v8f;

union AFrag { v16h h; uint32_t u[8]; uint4 q[2]; };

#define NWAVE 8   // waves per block
#define MPW   32  // points per wave

// K offset (within a 32-wide k-tile) held by dword v on `lane`
// (ISA 7.12.2: 16-bit 16x32 fragment layout; A and B assumed symmetric).
__device__ __forceinline__ int kof(int v, int lane) {
  return ((v >> 2) << 4) + ((lane >> 4) << 3) + ((v & 3) << 1);
}

__device__ __forceinline__ uint32_t pack2h(float a, float b) {
  union { _Float16 h[2]; uint32_t u; } c;
  c.h[0] = (_Float16)a; c.h[1] = (_Float16)b;
  return c.u;
}

// packed f16 relu: one VOP3P instruction for both halves
__device__ __forceinline__ uint32_t relu_pk(uint32_t x) {
  uint32_t r;
  const uint32_t z = 0;
  asm("v_pk_max_num_f16 %0, %1, %2" : "=v"(r) : "v"(x), "v"(z));
  return r;
}

__device__ __forceinline__ void lds_sync() {
  asm volatile("s_wait_dscnt 0" ::: "memory");
}

__device__ __forceinline__ float softplusf(float x) {
  return (x > 20.f) ? x : ((x < -20.f) ? __expf(x) : log1pf(__expf(x)));
}

// ---------------------------------------------------------------------------
// Prep: pack one f32 weight matrix (OUT x IN, row major) into f16 WMMA
// A-fragments: frag(kt,mt) = 32 lanes x 8 dwords; dword v of lane l holds the
// f16 pair (K = kt*32 + kof(v,l), K+1) of output-feature row mt*16 + (l&15).
// ---------------------------------------------------------------------------
__global__ void pack_weights_kernel(const float* __restrict__ W, int OUT, int IN,
                                    int MT, uint32_t* __restrict__ dst) {
  const int b = blockIdx.x;                 // fragment index = kt*MT + mt
  const int lane = threadIdx.x & 31, v = threadIdx.x >> 5;
  const int kt = b / MT, mt = b % MT;
  const int K = kt * 32 + kof(v, lane);
  const int M = mt * 16 + (lane & 15);
  float a = (M < OUT && K     < IN) ? W[M * IN + K    ] : 0.f;
  float c = (M < OUT && K + 1 < IN) ? W[M * IN + K + 1] : 0.f;
  dst[b * 256 + lane * 8 + v] = pack2h(a, c);
}

// fragment start offsets (u32 units) for the 14 weight layers in d_ws
constexpr int WOFF[14] = { 0, 2048, 4096, 4608, 6656, 8704, 10752,
                           11264, 14336, 16384, 17408, 19456, 21504, 23552 };

// ---------------------------------------------------------------------------
// denseT: h^T[mt-tile] = W * act^T, two point-tiles share every weight frag.
//   acc[mt][tile] ; D element = (feat mt*16 + r + 8*(lane/16), point lane%16)
// ---------------------------------------------------------------------------
template<int KT, int MT>
__device__ __forceinline__ void denseT(const uint32_t* __restrict__ Wf,
                                       const AFrag* B0, const AFrag* B1,
                                       v8f (*acc)[2], int lane) {
#pragma unroll
  for (int mt = 0; mt < MT; ++mt) {
    v8f c0 = {0.f, 0.f, 0.f, 0.f, 0.f, 0.f, 0.f, 0.f};
    v8f c1 = {0.f, 0.f, 0.f, 0.f, 0.f, 0.f, 0.f, 0.f};
#pragma unroll
    for (int kt = 0; kt < KT; ++kt) {
      AFrag wf;
      const uint4* p = (const uint4*)(Wf + (kt * MT + mt) * 256 + lane * 8);
      wf.q[0] = p[0];
      wf.q[1] = p[1];
      c0 = __builtin_amdgcn_wmma_f32_16x16x32_f16(false, wf.h, false, B0[kt].h,
                                                  (short)0, c0, false, false);
      c1 = __builtin_amdgcn_wmma_f32_16x16x32_f16(false, wf.h, false, B1[kt].h,
                                                  (short)0, c1, false, false);
    }
    acc[mt][0] = c0;
    acc[mt][1] = c1;
  }
}

// D (64 feats, 4 m-tiles) -> relu -> next-layer B fragments; LANE-LOCAL.
// Bnext[kt].u[v] must hold feats kt*32 + (v/4)*16 + 8*(lane/16) + (v%4)*2 (+1)
// of this lane's point == acc[2*kt + v/4][tile][(v%4)*2 (+1)].
__device__ __forceinline__ void repack_relu(const v8f (*acc)[2],
                                            AFrag* B0, AFrag* B1) {
#pragma unroll
  for (int kt = 0; kt < 2; ++kt) {
#pragma unroll
    for (int v = 0; v < 8; ++v) {
      const int ft = 2 * kt + (v >> 2);
      const int r  = (v & 3) * 2;
      B0[kt].u[v] = relu_pk(pack2h(acc[ft][0][r], acc[ft][0][r + 1]));
      B1[kt].u[v] = relu_pk(pack2h(acc[ft][1][r], acc[ft][1][r + 1]));
    }
  }
}

// color-net input feature c of this lane's point:
//   c in [0,27)  : view dirs, straight from the x row (global, L2-hot)
//   c in [27,42) : geo = sigma-net output feat (c-27+GOFF) via LDS buf
//   c >= 42     : zero pad
template<int GOFF>
__device__ __forceinline__ _Float16 cfeat(const float* __restrict__ xr,
                                          const _Float16* __restrict__ bp, int c) {
  if (c < 27) return (_Float16)xr[71 + c];
  if (c < 42) return bp[c - 27 + GOFF];
  return (_Float16)0.f;
}

template<int GOFF>
__device__ __forceinline__ void build_colorB(const float* __restrict__ xr,
                                             const _Float16* __restrict__ bp,
                                             AFrag* B, int lane) {
#pragma unroll
  for (int kt = 0; kt < 2; ++kt) {
#pragma unroll
    for (int v = 0; v < 8; ++v) {
      const int c = kt * 32 + kof(v, lane);
      union { _Float16 h[2]; uint32_t u; } pk;
      pk.h[0] = cfeat<GOFF>(xr, bp, c);
      pk.h[1] = cfeat<GOFF>(xr, bp, c + 1);
      B[kt].u[v] = pk.u;
    }
  }
}

// ---------------------------------------------------------------------------
// Main fused kernel: 8 waves/block, 32 points/wave (256 points/block).
// ---------------------------------------------------------------------------
__global__ __launch_bounds__(256)
void nerf_fused_kernel(const float* __restrict__ x, const uint32_t* __restrict__ Wp,
                       float* __restrict__ out, int npts) {
  const int lane = threadIdx.x & 31;
  const int w    = threadIdx.x >> 5;
  const int m    = lane & 15;        // this lane's point within each tile
  const int h    = lane >> 4;        // half-wave (feat sub-block 8h..8h+7)
  const long long base = (long long)blockIdx.x * (NWAVE * MPW) + (long long)w * MPW;

  long long pm0 = base + m;       if (pm0 >= npts) pm0 = npts - 1;
  long long pm1 = base + 16 + m;  if (pm1 >= npts) pm1 = npts - 1;
  const float* xr0 = x + pm0 * 98;
  const float* xr1 = x + pm1 * 98;

  // sigma-net raw outputs (17 feats max) per tile/point, f16
  __shared__ __align__(8)  _Float16 bufS[NWAVE][2][16][24];
  __shared__ __align__(16) float    osS[NWAVE][MPW * 12];
  _Float16 (*buf)[16][24] = bufS[w];
  float* os = osS[w];

  // ---- first-layer B fragments straight from x (f32 -> f16), two tiles ----
  AFrag fgB0[3], fgB1[3];                   // K = 0..95, zero past 70
#pragma unroll
  for (int kt = 0; kt < 3; ++kt) {
#pragma unroll
    for (int v = 0; v < 8; ++v) {
      const int K = kt * 32 + kof(v, lane);
      float a0 = (K     < 71) ? xr0[K]     : 0.f;
      float b0 = (K + 1 < 71) ? xr0[K + 1] : 0.f;
      float a1 = (K     < 71) ? xr1[K]     : 0.f;
      float b1 = (K + 1 < 71) ? xr1[K + 1] : 0.f;
      fgB0[kt].u[v] = pack2h(a0, b0);
      fgB1[kt].u[v] = pack2h(a1, b1);
    }
  }

  v8f acc[4][2];
  AFrag B0[2], B1[2];

  // ======================= FG sigma MLP =======================
  denseT<3, 4>(Wp + WOFF[7], fgB0, fgB1, acc, lane);
  repack_relu(acc, B0, B1);
  denseT<2, 4>(Wp + WOFF[8], B0, B1, acc, lane);
  repack_relu(acc, B0, B1);
  denseT<2, 2>(Wp + WOFF[9], B0, B1, acc, lane);   // 17 raw outs, 2 m-tiles
  // lane (m,h) owns s2 feats {8h..8h+7} (mt0) and {16+8h..} (mt1, feat16 only)
#pragma unroll
  for (int t = 0; t < 2; ++t) {
#pragma unroll
    for (int r = 0; r < 8; ++r)
      buf[t][m][r + 8 * h] = (_Float16)acc[0][t][r];
    if (h == 0) {
      buf[t][m][16] = (_Float16)acc[1][t][0];       // geo[14]
      os[(t * 16 + m) * 12 + 4] = acc[0][t][0];     // fg sigma raw
      os[(t * 16 + m) * 12 + 5] = acc[0][t][1];     // unc raw
    }
  }
  lds_sync();

  // ======================= FG color MLP =======================
  build_colorB<2>(xr0, &buf[0][m][0], B0, lane);    // fg geo = s2 feats 2..16
  build_colorB<2>(xr1, &buf[1][m][0], B1, lane);
  denseT<2, 4>(Wp + WOFF[10], B0, B1, acc, lane);
  repack_relu(acc, B0, B1);
  denseT<2, 4>(Wp + WOFF[11], B0, B1, acc, lane);
  repack_relu(acc, B0, B1);
  denseT<2, 4>(Wp + WOFF[12], B0, B1, acc, lane);
  repack_relu(acc, B0, B1);
  denseT<2, 1>(Wp + WOFF[13], B0, B1, acc, lane);   // 3 outs: h=0, r=0..2
  if (h == 0) {
#pragma unroll
    for (int t = 0; t < 2; ++t)
#pragma unroll
      for (int r = 0; r < 3; ++r)
        os[(t * 16 + m) * 12 + 6 + r] = acc[0][t][r];
  }

  // ---- fg input fragments are dead: mask in place -> bg input ----
  // bg drops the 8 time dims (only K=0..62): zero feature 63
  // (k-tile 1, lanes >= 16, dword 7 high half); reuse k-tiles 0..1.
  if (lane >= 16) { fgB0[1].u[7] &= 0x0000FFFFu; fgB1[1].u[7] &= 0x0000FFFFu; }

  // ======================= BG sigma MLP =======================
  denseT<2, 4>(Wp + WOFF[0], fgB0, fgB1, acc, lane);
  repack_relu(acc, B0, B1);
  denseT<2, 4>(Wp + WOFF[1], B0, B1, acc, lane);
  repack_relu(acc, B0, B1);
  denseT<2, 1>(Wp + WOFF[2], B0, B1, acc, lane);    // 16 raw outs
#pragma unroll
  for (int t = 0; t < 2; ++t) {
#pragma unroll
    for (int r = 0; r < 8; ++r)
      buf[t][m][r + 8 * h] = (_Float16)acc[0][t][r];
    if (h == 0) os[(t * 16 + m) * 12 + 0] = acc[0][t][0];  // bg sigma raw
  }
  lds_sync();

  // ======================= BG color MLP =======================
  build_colorB<1>(xr0, &buf[0][m][0], B0, lane);    // bg geo = s2 feats 1..15
  build_colorB<1>(xr1, &buf[1][m][0], B1, lane);
  denseT<2, 4>(Wp + WOFF[3], B0, B1, acc, lane);
  repack_relu(acc, B0, B1);
  denseT<2, 4>(Wp + WOFF[4], B0, B1, acc, lane);
  repack_relu(acc, B0, B1);
  denseT<2, 4>(Wp + WOFF[5], B0, B1, acc, lane);
  repack_relu(acc, B0, B1);
  denseT<2, 1>(Wp + WOFF[6], B0, B1, acc, lane);
  if (h == 0) {
#pragma unroll
    for (int t = 0; t < 2; ++t)
#pragma unroll
      for (int r = 0; r < 3; ++r)
        os[(t * 16 + m) * 12 + 1 + r] = acc[0][t][r];
  }
  lds_sync();

  // ======================= epilogue (one point per lane) ======
  if (base + lane < npts) {
    const float sb = softplusf(os[lane * 12 + 0]);
    const float sf = softplusf(os[lane * 12 + 4]);
    const float un = softplusf(os[lane * 12 + 5]);
    const float sg = sb + sf + 1e-9f;
    const float wb = sb / sg, wf = sf / sg;
    float* o = out + (base + lane) * 6;
    o[0] = wb * os[lane * 12 + 1] + wf * os[lane * 12 + 6];
    o[1] = wb * os[lane * 12 + 2] + wf * os[lane * 12 + 7];
    o[2] = wb * os[lane * 12 + 3] + wf * os[lane * 12 + 8];
    o[3] = sg;
    o[4] = un;
    o[5] = sf;
  }
}

// ---------------------------------------------------------------------------
extern "C" void kernel_launch(void* const* d_in, const int* in_sizes, int n_in,
                              void* d_out, int out_size, void* d_ws, size_t ws_size,
                              hipStream_t stream) {
  (void)n_in; (void)out_size; (void)ws_size;   // needs 96,256 B of d_ws
  // layer order matches d_in[1..14]: bg_s0..bg_c3, fg_s0..fg_c3
  static const int OUTD[14] = {64, 64, 16, 64, 64, 64, 3, 64, 64, 17, 64, 64, 64, 3};
  static const int IND [14] = {63, 64, 64, 42, 64, 64, 64, 71, 64, 64, 42, 64, 64, 64};
  static const int MT  [14] = {4, 4, 1, 4, 4, 4, 1, 4, 4, 2, 4, 4, 4, 1};
  static const int KT  [14] = {2, 2, 2, 2, 2, 2, 2, 3, 2, 2, 2, 2, 2, 2};
  static const int OFFH[14] = {0, 2048, 4096, 4608, 6656, 8704, 10752,
                               11264, 14336, 16384, 17408, 19456, 21504, 23552};

  uint32_t* wp = (uint32_t*)d_ws;
  for (int i = 0; i < 14; ++i) {
    pack_weights_kernel<<<MT[i] * KT[i], 256, 0, stream>>>(
        (const float*)d_in[1 + i], OUTD[i], IND[i], MT[i], wp + OFFH[i]);
  }

  const int npts = in_sizes[0] / 98;            // rows of x (IN_CH + IN_VIEWS = 98)
  const int nblocks = (npts + NWAVE * MPW - 1) / (NWAVE * MPW);  // 256 pts/block
  nerf_fused_kernel<<<nblocks, 256, 0, stream>>>(
      (const float*)d_in[0], wp, (float*)d_out, npts);
}
